// ENLTransformerBlock_21380347199745
// MI455X (gfx1250) — matrix operand order repaired and checked by reading, not compiled, not run on hardware
//
#include <hip/hip_runtime.h>
#include <hip/hip_bf16.h>

typedef __attribute__((ext_vector_type(16))) _Float16 v16h;
typedef __attribute__((ext_vector_type(8)))  float    v8f;

#define DIMC   256
#define NTOK   4096
#define BATCH  8
#define HIDDEN 1024
#define APAD   72   // halves per LDS A-row: 64 data + 8 pad -> 144B (16B aligned, bank-conflict-free)

// ---------------------------------------------------------------------------
// WMMA helpers (CDNA5 gfx1250, wave32)
// ---------------------------------------------------------------------------
static __device__ __forceinline__ v8f wmma16x16x32(v16h a, v16h b, v8f c) {
  return __builtin_amdgcn_wmma_f32_16x16x32_f16(false, a, false, b, (short)0, c,
                                                false, false);
}

// A fragment 16x32 f16 read from LDS tile (row stride APAD halves), K-chunk kk.
// ISA layout: lane<16 -> M=lane, K {0..7} in v0..3, {16..23} in v4..7
//             lane>=16 -> M=lane-16, K {8..15} and {24..31}
static __device__ __forceinline__ v16h lds_a_frag(const _Float16* sm, int m0,
                                                  int kk) {
  const int lane = threadIdx.x & 31;
  const int row  = m0 + (lane & 15);
  const int kb   = kk + ((lane & 16) ? 8 : 0);
  const _Float16* p = sm + row * APAD + kb;
  union { v16h h; uint4 u[2]; } f;
  f.u[0] = *(const uint4*)(p);        // k = kb .. kb+7   (16B)
  f.u[1] = *(const uint4*)(p + 16);   // k = kb+16 .. kb+23
  return f.h;
}

// B fragment: 32x16 f16, from W row-major [N,K] (B = W^T), direct from global.
// lane<16 -> N=lane, K {0..7},{16..23}; lane>=16 -> N=lane-16, K {8..15},{24..31}
static __device__ __forceinline__ v16h load_b_frag(const _Float16* W, int ldw,
                                                   int n0, int k0) {
  const int lane = threadIdx.x & 31;
  const int col  = n0 + (lane & 15);
  const int kb   = k0 + ((lane & 16) ? 8 : 0);
  const _Float16* p = W + (long long)col * ldw + kb;
  union { v16h h; uint4 u[2]; } f;
  f.u[0] = *(const uint4*)(p);
  f.u[1] = *(const uint4*)(p + 16);
  return f.h;
}

// gfx1250 async copy: global -> LDS, 16 bytes per lane, tracked by ASYNCcnt.
static __device__ __forceinline__ void async_copy_b128(uint32_t lds_byte_addr,
                                                       const void* gaddr) {
  asm volatile("global_load_async_to_lds_b128 %0, %1, off"
               :: "v"(lds_byte_addr), "v"((unsigned long long)(uintptr_t)gaddr)
               : "memory");
}
static __device__ __forceinline__ void wait_asynccnt0() {
  asm volatile("s_wait_asynccnt 0" ::: "memory");
}

// ---------------------------------------------------------------------------
// Tiled GEMM:  D[M,N] = A[M,K] @ W[N,K]^T  (+ epilogue)
// block = 256 threads (8 waves). block tile 64(M) x 128(N). K-step 64.
// A tile double-buffered in LDS via global_load_async_to_lds_b128; last K-step
// peeled so the in-loop staging is unconditional (branch-free hot loop).
// EPI: 0 = +bias
//      1 = FAVOR feature map: ratio*(exp(acc/max(rn,5e-5) - 0.5*rn^2/rh^2)+1e-4)
//      2 = +bias then exact GELU
//      3 = acc*dinv[m] + resid[m,n]      (fp32 out, may alias resid)
//      4 = acc + bias + resid[m,n]       (fp32 out)
// ---------------------------------------------------------------------------
template <int EPI, bool TRANS, bool F16OUT>
__global__ __launch_bounds__(256)
void gemm_wmma(const _Float16* A, long long sA, int lda,
               const _Float16* W, long long sW, int ldw,
               void* O, long long sO, int ldo,
               const float* bias,
               const float* rowaux, long long sAux,
               const float* resid, long long sRes, int ldres,
               int K, float ratio) {
  const int bz = blockIdx.z;
  A += (long long)bz * sA;
  W += (long long)bz * sW;
  if (rowaux) rowaux += (long long)bz * sAux;
  if (resid)  resid  += (long long)bz * sRes;

  __shared__ __align__(16) _Float16 smA[2][64 * APAD];

  const int m_blk  = blockIdx.y * 64;
  const int wave   = threadIdx.x >> 5;
  const int m_loc  = (wave & 1) * 32;                 // wave M-offset within tile
  const int n_base = blockIdx.x * 128 + (wave >> 1) * 32;

  // cooperative async stage of A[m_blk..+63][k0..+63] into smA[buf]:
  // 512 chunks of 16B; each thread copies 2 chunks (rows r and r+32).
  const int srow = threadIdx.x >> 3;        // 0..31
  const int soff = (threadIdx.x & 7) * 8;   // 0..56 halves
  const _Float16* sgp0 = A + (long long)(m_blk + srow) * lda + soff;
  const _Float16* sgp1 = A + (long long)(m_blk + srow + 32) * lda + soff;
  const uint32_t sl0 = (uint32_t)(uintptr_t)&smA[0][srow * APAD + soff];
  const uint32_t sl1 = (uint32_t)(uintptr_t)&smA[0][(srow + 32) * APAD + soff];
  const uint32_t lbuf = (uint32_t)(64 * APAD * sizeof(_Float16));
  auto stage_A = [&](int buf, int k0) {
    async_copy_b128(sl0 + buf * lbuf, (const void*)(sgp0 + k0));
    async_copy_b128(sl1 + buf * lbuf, (const void*)(sgp1 + k0));
  };

  v8f acc[2][2] = {};
  auto compute = [&](const _Float16* sm, int k0) {
#pragma unroll
    for (int kk = 0; kk < 64; kk += 32) {
      v16h a0 = lds_a_frag(sm, m_loc,      kk);
      v16h a1 = lds_a_frag(sm, m_loc + 16, kk);
      v16h b0 = load_b_frag(W, ldw, n_base,      k0 + kk);
      v16h b1 = load_b_frag(W, ldw, n_base + 16, k0 + kk);
      acc[0][0] = wmma16x16x32(a0, b0, acc[0][0]);
      acc[0][1] = wmma16x16x32(a0, b1, acc[0][1]);
      acc[1][0] = wmma16x16x32(a1, b0, acc[1][0]);
      acc[1][1] = wmma16x16x32(a1, b1, acc[1][1]);
    }
  };

  stage_A(0, 0);
  wait_asynccnt0();
  __syncthreads();

  int buf = 0;
  int k0 = 0;
  for (; k0 + 64 < K; k0 += 64) {      // branch-free hot loop
    stage_A(buf ^ 1, k0 + 64);
    compute(&smA[buf][0], k0);
    wait_asynccnt0();
    __syncthreads();
    buf ^= 1;
  }
  compute(&smA[buf][0], k0);           // peeled last K-step

  // C/D layout: lane<16 -> vgpr r = (M=r, N=lane); lane>=16 -> (M=8+r, N=lane-16)
  const int lane = threadIdx.x & 31;
  const int hi   = lane >> 4;
  const int colo = lane & 15;

  for (int i = 0; i < 2; ++i) {
    for (int j = 0; j < 2; ++j) {
      const int n = n_base + 16 * j + colo;
      float bv = 0.f;
      if constexpr (EPI == 0 || EPI == 2 || EPI == 4) {
        if (bias) bv = bias[n];
      }
      for (int r = 0; r < 8; ++r) {
        const int m = m_blk + m_loc + 16 * i + hi * 8 + r;
        float v = acc[i][j][r];
        if constexpr (EPI == 0) {
          v += bv;
        } else if constexpr (EPI == 1) {
          const float rn = rowaux[m];
          const float rh = fmaxf(rn, 5e-5f);
          v = ratio * (expf(v / rh - 0.5f * (rn * rn) / (rh * rh)) + 1e-4f);
        } else if constexpr (EPI == 2) {
          v += bv;
          v = 0.5f * v * (1.0f + erff(v * 0.70710678118654752f));
        } else if constexpr (EPI == 3) {
          v = v * rowaux[m] + resid[(long long)m * ldres + n];
        } else if constexpr (EPI == 4) {
          v += bv + resid[(long long)m * ldres + n];
        }
        const long long idx =
            TRANS ? ((long long)n * ldo + m) : ((long long)m * ldo + n);
        if constexpr (F16OUT)
          ((_Float16*)O)[(long long)bz * sO + idx] = (_Float16)v;
        else
          ((float*)O)[(long long)bz * sO + idx] = v;
      }
    }
  }
}

// ---------------------------------------------------------------------------
// Reduction helpers / small kernels
// ---------------------------------------------------------------------------
static __device__ __forceinline__ float block_sum_256(float v, float* smem) {
  const int t = threadIdx.x;
  smem[t] = v;
  __syncthreads();
  for (int off = 128; off > 0; off >>= 1) {
    if (t < off) smem[t] += smem[t + off];
    __syncthreads();
  }
  float r = smem[0];
  __syncthreads();
  return r;
}

__global__ void convert_f32_f16(const float* in, _Float16* out, int n) {
  int i = blockIdx.x * blockDim.x + threadIdx.x;
  if (i < n) out[i] = (_Float16)in[i];
}

// One block (256 threads) per token. Generic strided input:
// in[b*bs + n*ts + c*cs].  Writes h_out f16 token-major and optional t_out f32.
__global__ void token_layernorm(const float* in, long long bs, long long ts,
                                long long cs, const float* g, const float* beta,
                                float* t_out, _Float16* h_out) {
  const int n = blockIdx.x, b = blockIdx.y, c = threadIdx.x;
  const long long base = (long long)b * bs + (long long)n * ts;
  const float v = in[base + (long long)c * cs];
  __shared__ float sm[256];
  const float s  = block_sum_256(v, sm);
  const float ss = block_sum_256(v * v, sm);
  const float mu  = s * (1.0f / 256.0f);
  const float var = ss * (1.0f / 256.0f) - mu * mu;
  const float rs  = rsqrtf(var + 1e-5f);
  const long long oi = ((long long)b * gridDim.x + n) * DIMC + c;
  h_out[oi] = (_Float16)((v - mu) * rs * g[c] + beta[c]);
  if (t_out) t_out[oi] = v;
}

// rnorm[b,n] = ||row||_2 over C=256 f16 values
__global__ void rownorm_kernel(const _Float16* q, float* out) {
  const int n = blockIdx.x, b = blockIdx.y, c = threadIdx.x;
  const float v = (float)q[((long long)b * gridDim.x + n) * DIMC + c];
  __shared__ float sm[256];
  const float s = block_sum_256(v * v, sm);
  if (c == 0) out[(long long)b * gridDim.x + n] = sqrtf(s);
}

// ksum[b,f] = sum_n kpT[b,f,n]
__global__ void ksum_kernel(const _Float16* kpT, float* ksum, int ntok) {
  const int f = blockIdx.x, b = blockIdx.y, t = threadIdx.x;
  const _Float16* p = kpT + ((long long)b * gridDim.x + f) * ntok;
  float s = 0.f;
  for (int i = t; i < ntok; i += 256) s += (float)p[i];
  __shared__ float sm[256];
  s = block_sum_256(s, sm);
  if (t == 0) ksum[(long long)b * gridDim.x + f] = s;
}

// dinv[b,n] = 1 / (qp[b,n,:] . ksum[b,:])
__global__ void dinv_kernel(const _Float16* qp, const float* ksum, float* dinv) {
  const int n = blockIdx.x, b = blockIdx.y, f = threadIdx.x;
  const float v =
      (float)qp[((long long)b * gridDim.x + n) * DIMC + f] * ksum[b * DIMC + f];
  __shared__ float sm[256];
  const float s = block_sum_256(v, sm);
  if (f == 0) dinv[(long long)b * gridDim.x + n] = 1.0f / s;
}

// ---------------------------------------------------------------------------
extern "C" void kernel_launch(void* const* d_in, const int* in_sizes, int n_in,
                              void* d_out, int out_size, void* d_ws,
                              size_t ws_size, hipStream_t stream) {
  (void)in_sizes; (void)n_in; (void)out_size; (void)ws_size;
  const float* x    = (const float*)d_in[0];
  const float* proj = (const float*)d_in[1];
  const float* wq = (const float*)d_in[2];   const float* bq  = (const float*)d_in[3];
  const float* wk = (const float*)d_in[4];   const float* bk  = (const float*)d_in[5];
  const float* wa = (const float*)d_in[6];   const float* ba  = (const float*)d_in[7];
  const float* g1 = (const float*)d_in[8];   const float* b1  = (const float*)d_in[9];
  const float* g2 = (const float*)d_in[10];  const float* b2  = (const float*)d_in[11];
  const float* w1 = (const float*)d_in[12];  const float* fb1 = (const float*)d_in[13];
  const float* w2 = (const float*)d_in[14];  const float* fb2 = (const float*)d_in[15];

  // ---- workspace layout (bump allocator, 256B aligned) ----
  char* wsp = (char*)d_ws;
  auto alloc = [&](size_t bytes) -> void* {
    void* p = (void*)wsp;
    wsp += (bytes + 255) & ~(size_t)255;
    return p;
  };
  const long long TC = (long long)NTOK * DIMC;
  float*    tbuf = (float*)alloc((size_t)BATCH * TC * 4);      // shortcut / t2 (in-place)
  _Float16* h1   = (_Float16*)alloc((size_t)BATCH * TC * 2);   // LN1 out; reused for LN2 out
  _Float16* qb   = (_Float16*)alloc((size_t)BATCH * TC * 2);   // q; reused for kpT
  _Float16* kb_  = (_Float16*)alloc((size_t)BATCH * TC * 2);   // k
  _Float16* vT   = (_Float16*)alloc((size_t)BATCH * TC * 2);   // v transposed [C,Ntok]
  _Float16* qp   = (_Float16*)alloc((size_t)BATCH * TC * 2);   // FAVOR(q)
  _Float16* m1   = (_Float16*)alloc((size_t)BATCH * NTOK * HIDDEN * 2);
  _Float16* ctxT = (_Float16*)alloc((size_t)BATCH * DIMC * DIMC * 2);
  float* rq    = (float*)alloc((size_t)BATCH * NTOK * 4);
  float* rk    = (float*)alloc((size_t)BATCH * NTOK * 4);
  float* dinvb = (float*)alloc((size_t)BATCH * NTOK * 4);
  float* ksumb = (float*)alloc((size_t)BATCH * DIMC * 4);
  _Float16* wqh   = (_Float16*)alloc((size_t)DIMC * DIMC * 2);
  _Float16* wkh   = (_Float16*)alloc((size_t)DIMC * DIMC * 2);
  _Float16* wah   = (_Float16*)alloc((size_t)DIMC * DIMC * 2);
  _Float16* projh = (_Float16*)alloc((size_t)DIMC * DIMC * 2);
  _Float16* w1h   = (_Float16*)alloc((size_t)HIDDEN * DIMC * 2);
  _Float16* w2h   = (_Float16*)alloc((size_t)DIMC * HIDDEN * 2);
  _Float16* kpT   = qb;   // reuse: q is dead after FAVOR(q)
  _Float16* h2    = h1;   // reuse: h1 is dead after q/k/v GEMMs

  const dim3 blk(256);
  const float ratio = 0.0625f;  // 256^-0.5

  // 1) weights -> f16
  convert_f32_f16<<<DIMC * DIMC / 256, blk, 0, stream>>>(wq, wqh, DIMC * DIMC);
  convert_f32_f16<<<DIMC * DIMC / 256, blk, 0, stream>>>(wk, wkh, DIMC * DIMC);
  convert_f32_f16<<<DIMC * DIMC / 256, blk, 0, stream>>>(wa, wah, DIMC * DIMC);
  convert_f32_f16<<<DIMC * DIMC / 256, blk, 0, stream>>>(proj, projh, DIMC * DIMC);
  convert_f32_f16<<<HIDDEN * DIMC / 256, blk, 0, stream>>>(w1, w1h, HIDDEN * DIMC);
  convert_f32_f16<<<DIMC * HIDDEN / 256, blk, 0, stream>>>(w2, w2h, DIMC * HIDDEN);

  // 2) tokens + LN1: x[b,c,n] -> t fp32 + h1 f16  (token-major)
  token_layernorm<<<dim3(NTOK, BATCH), blk, 0, stream>>>(
      x, (long long)DIMC * NTOK, 1LL, (long long)NTOK, g1, b1, tbuf, h1);

  // 3) q/k/v projections (v stored transposed [C,Ntok])
  gemm_wmma<0, false, true><<<dim3(DIMC / 128, NTOK / 64, BATCH), blk, 0, stream>>>(
      h1, TC, DIMC, wqh, 0LL, DIMC, qb, TC, DIMC, bq,
      nullptr, 0LL, nullptr, 0LL, 0, DIMC, 0.f);
  gemm_wmma<0, false, true><<<dim3(DIMC / 128, NTOK / 64, BATCH), blk, 0, stream>>>(
      h1, TC, DIMC, wkh, 0LL, DIMC, kb_, TC, DIMC, bk,
      nullptr, 0LL, nullptr, 0LL, 0, DIMC, 0.f);
  gemm_wmma<0, true, true><<<dim3(DIMC / 128, NTOK / 64, BATCH), blk, 0, stream>>>(
      h1, TC, DIMC, wah, 0LL, DIMC, vT, TC, NTOK, ba,
      nullptr, 0LL, nullptr, 0LL, 0, DIMC, 0.f);

  // 4) row L2 norms of q and k
  rownorm_kernel<<<dim3(NTOK, BATCH), blk, 0, stream>>>(qb, rq);
  rownorm_kernel<<<dim3(NTOK, BATCH), blk, 0, stream>>>(kb_, rk);

  // 5) FAVOR+ feature maps (normalize fused via row norms)
  gemm_wmma<1, false, true><<<dim3(DIMC / 128, NTOK / 64, BATCH), blk, 0, stream>>>(
      qb, TC, DIMC, projh, 0LL, DIMC, qp, TC, DIMC, nullptr,
      rq, NTOK, nullptr, 0LL, 0, DIMC, ratio);
  gemm_wmma<1, true, true><<<dim3(DIMC / 128, NTOK / 64, BATCH), blk, 0, stream>>>(
      kb_, TC, DIMC, projh, 0LL, DIMC, kpT, TC, NTOK, nullptr,
      rk, NTOK, nullptr, 0LL, 0, DIMC, ratio);

  // 6) k_sum and D_inv
  ksum_kernel<<<dim3(DIMC, BATCH), blk, 0, stream>>>(kpT, ksumb, NTOK);
  dinv_kernel<<<dim3(NTOK, BATCH), blk, 0, stream>>>(qp, ksumb, dinvb);

  // 7) ctx^T[c,f] = sum_n v[n,c] kp[n,f]   (A=vT, W=kpT, K=Ntok)
  gemm_wmma<0, false, true><<<dim3(DIMC / 128, DIMC / 64, BATCH), blk, 0, stream>>>(
      vT, TC, NTOK, kpT, TC, NTOK, ctxT, (long long)DIMC * DIMC, DIMC, nullptr,
      nullptr, 0LL, nullptr, 0LL, 0, NTOK, 0.f);

  // 8) out = (qp @ ctx) * D_inv + shortcut  -> t2 (in-place over tbuf, fp32)
  gemm_wmma<3, false, false><<<dim3(DIMC / 128, NTOK / 64, BATCH), blk, 0, stream>>>(
      qp, TC, DIMC, ctxT, (long long)DIMC * DIMC, DIMC, tbuf, TC, DIMC, nullptr,
      dinvb, NTOK, tbuf, TC, DIMC, DIMC, 0.f);

  // 9) LN2 on t2 (token-major fp32) -> h2 f16
  token_layernorm<<<dim3(NTOK, BATCH), blk, 0, stream>>>(
      tbuf, TC, (long long)DIMC, 1LL, g2, b2, nullptr, h2);

  // 10) MLP up + exact GELU
  gemm_wmma<2, false, true><<<dim3(HIDDEN / 128, NTOK / 64, BATCH), blk, 0, stream>>>(
      h2, TC, DIMC, w1h, 0LL, DIMC, m1, (long long)NTOK * HIDDEN, HIDDEN, fb1,
      nullptr, 0LL, nullptr, 0LL, 0, DIMC, 0.f);

  // 11) MLP down + residual, store transposed fp32 directly as [B,C,H,W]
  gemm_wmma<4, true, false><<<dim3(DIMC / 128, NTOK / 64, BATCH), blk, 0, stream>>>(
      m1, (long long)NTOK * HIDDEN, HIDDEN, w2h, 0LL, HIDDEN,
      d_out, (long long)DIMC * NTOK, NTOK, fb2,
      nullptr, 0LL, tbuf, TC, DIMC, HIDDEN, 0.f);
}
